// PartialRunnerForMambaLMHeadModel_56083682951934
// MI455X (gfx1250) — matrix operand order repaired and checked by reading, not compiled, not run on hardware
//
#include <hip/hip_runtime.h>
#include <hip/hip_bf16.h>

// ---------------------------------------------------------------------------
// Mamba forward for MI455X (gfx1250, wave32, WMMA).
// All GEMMs use v_wmma_f32_16x16x32_bf16 with fused epilogues and a
// software-pipelined (double-buffered) K-loop for latency hiding.
// ---------------------------------------------------------------------------

typedef __bf16 bf16;
typedef __attribute__((ext_vector_type(16))) __bf16 v16bf;
typedef __attribute__((ext_vector_type(8)))  __bf16 v8bf;
typedef __attribute__((ext_vector_type(8)))  float  v8f;

#define D_MODEL 768
#define D_INNER 1536
#define D_XZ    3072      // 2*D_INNER
#define D_STATE 16
#define DT_RANK 48
#define DT_PAD  64        // DT_RANK padded to WMMA K granularity
#define D_CONV  4
#define SEQ     1024
#define N_LAYER 4
#define XDBC    80        // DT_RANK + 2*D_STATE
#define EPS     1e-5f

__device__ __forceinline__ v8f wmma_bf16(v16bf a, v16bf b, v8f c) {
#if defined(__gfx1250__)
    return __builtin_amdgcn_wmma_f32_16x16x32_bf16(
        /*neg_a=*/false, a, /*neg_b=*/false, b,
        /*c_mod=*/(short)0, c, /*reuse_a=*/false, /*reuse_b=*/false);
#else
    (void)a; (void)b;
    return c;   // host-pass placeholder, never executed on device
#endif
}

__device__ __forceinline__ float silu_f(float x) {
    return x / (1.0f + expf(-x));
}
__device__ __forceinline__ float softplus_f(float x) {
    return (x > 20.0f) ? x : log1pf(expf(x));
}

__device__ __forceinline__ v16bf cat8(v8bf lo, v8bf hi) {
    return __builtin_shufflevector(lo, hi,
        0,1,2,3,4,5,6,7,8,9,10,11,12,13,14,15);
}

// ---------------------------------------------------------------------------
// WMMA GEMM:  C[M x N] = A[M x K] (bf16, row-major) * W[N x K]^T (bf16, row-major)
// Wave computes a 16 x (NT*16) strip. 8 waves/block -> 128 rows per block.
// K-loop is double-buffered: next-step fragments are fetched before the
// current step's WMMAs so loads overlap the matrix pipe.
// EPI: 0 = plain store, 1 = += extra[r*ldc+n] (residual, may alias C),
//      2 = softplus(acc + extra[n]) (dt_proj bias+activation).
// ---------------------------------------------------------------------------
template <int NT, int EPI>
__global__ void gemm_wmma_kernel(const bf16* __restrict__ A,
                                 const bf16* __restrict__ W,
                                 float* C, const float* extra,
                                 int M, int N, int K,
                                 int lda, int ldw, int ldc) {
    const int lane  = threadIdx.x & 31;
    const int wave  = threadIdx.x >> 5;
    const int m0    = (blockIdx.y * (blockDim.x >> 5) + wave) * 16;
    const int nbase = blockIdx.x * (NT * 16);
    if (m0 >= M) return;

    const int half = lane >> 4;          // which 16-lane half of the wave
    const int l15  = lane & 15;

    // A fragment: per-lane chunks at k0 + 8*half and k0 + 16 + 8*half
    const bf16* Abase = A + (size_t)(m0 + l15) * lda + 8 * half;
    // B fragment: per-lane chunks of weight row n at k0 + 8*half and k0 + 16 + 8*half
    const bf16* Wbase[NT];
#pragma unroll
    for (int t = 0; t < NT; ++t)
        Wbase[t] = W + (size_t)(nbase + t * 16 + l15) * ldw + 8 * half;

    auto loadA = [&](int k0) -> v16bf {
        v8bf lo = *(const v8bf*)(Abase + k0);
        v8bf hi = *(const v8bf*)(Abase + k0 + 16);
        return cat8(lo, hi);
    };
    auto loadB = [&](int t, int k0) -> v16bf {
        v8bf lo = *(const v8bf*)(Wbase[t] + k0);
        v8bf hi = *(const v8bf*)(Wbase[t] + k0 + 16);
        return cat8(lo, hi);
    };

    v8f acc[NT];
#pragma unroll
    for (int t = 0; t < NT; ++t) acc[t] = (v8f){0,0,0,0,0,0,0,0};

    // ---- software-pipelined K loop ----
    v16bf a = loadA(0);
    v16bf b[NT];
#pragma unroll
    for (int t = 0; t < NT; ++t) b[t] = loadB(t, 0);

    for (int k0 = 0; k0 < K; k0 += 32) {
        // Next-step fetch (k=0 dummy on the final trip keeps this branch-free
        // and in-bounds; the fetched values are never consumed).
        const int kn = (k0 + 32 < K) ? (k0 + 32) : 0;
        __builtin_prefetch(Abase + kn + 32, 0, 3);      // near-cache prefetch
        v16bf a2 = loadA(kn);
        v16bf b2[NT];
#pragma unroll
        for (int t = 0; t < NT; ++t) b2[t] = loadB(t, kn);

        // Current-step WMMAs (depend only on a/b, overlap with fetches above)
#pragma unroll
        for (int t = 0; t < NT; ++t) acc[t] = wmma_bf16(a, b[t], acc[t]);

        a = a2;
#pragma unroll
        for (int t = 0; t < NT; ++t) b[t] = b2[t];
    }

    // D layout: element v -> row m0 + v + 8*half, col nbase + t*16 + (lane&15)
    const int rbase = m0 + 8 * half;
#pragma unroll
    for (int t = 0; t < NT; ++t) {
        const int n = nbase + t * 16 + l15;
#pragma unroll
        for (int v = 0; v < 8; ++v) {
            const int r = rbase + v;
            float val = acc[t][v];
            if (EPI == 1) val += extra[(size_t)r * ldc + n];
            if (EPI == 2) val = softplus_f(val + extra[n]);
            C[(size_t)r * ldc + n] = val;
        }
    }
}

// ---------------------------------------------------------------------------
// RMSNorm over last dim (D), one block per row, templated output type.
// ---------------------------------------------------------------------------
template <typename OutT>
__global__ void rmsnorm_kernel(const float* __restrict__ x,
                               const float* __restrict__ w,
                               OutT* __restrict__ out, int D) {
    const int row = blockIdx.x;
    const float* xr = x + (size_t)row * D;
    float ss = 0.0f;
    for (int i = threadIdx.x; i < D; i += blockDim.x) {
        float v = xr[i];
        ss += v * v;
    }
    __shared__ float red[256];
    red[threadIdx.x] = ss;
    __syncthreads();
    for (int off = 128; off > 0; off >>= 1) {
        if ((int)threadIdx.x < off) red[threadIdx.x] += red[threadIdx.x + off];
        __syncthreads();
    }
    const float rstd = rsqrtf(red[0] / (float)D + EPS);
    for (int i = threadIdx.x; i < D; i += blockDim.x)
        out[(size_t)row * D + i] = (OutT)(xr[i] * rstd * w[i]);
}

// ---------------------------------------------------------------------------
// Weight conversion kernels.
// ---------------------------------------------------------------------------
__global__ void f32_to_bf16_kernel(const float* __restrict__ src,
                                   bf16* __restrict__ dst, int n) {
    int i = blockIdx.x * blockDim.x + threadIdx.x;
    if (i < n) dst[i] = (bf16)src[i];
}

// dt_proj_w [rows x 48] -> [rows x 64] zero-padded bf16
__global__ void dtw_pad_kernel(const float* __restrict__ src,
                               bf16* __restrict__ dst, int n /* rows*64 */) {
    int i = blockIdx.x * blockDim.x + threadIdx.x;
    if (i >= n) return;
    int r = i >> 6, c = i & 63;
    dst[i] = (bf16)((c < DT_RANK) ? src[r * DT_RANK + c] : 0.0f);
}

// xdbc[:, :48] -> bf16 [SEQ x 64] zero-padded (A operand of dt_proj GEMM)
__global__ void dt_pad_act_kernel(const float* __restrict__ xdbc,
                                  bf16* __restrict__ dtpad) {
    int i = blockIdx.x * blockDim.x + threadIdx.x;   // SEQ*64
    int t = i >> 6, c = i & 63;
    dtpad[i] = (bf16)((c < DT_RANK) ? xdbc[t * XDBC + c] : 0.0f);
}

// ---------------------------------------------------------------------------
// Depthwise causal conv (k=4) + bias + SiLU. Reads xi half of xz.
// Emits f32 (for scan) and bf16 (for x_proj GEMM A operand).
// ---------------------------------------------------------------------------
__global__ void conv_silu_kernel(const float* __restrict__ xz,
                                 const float* __restrict__ cw,
                                 const float* __restrict__ cb,
                                 float* __restrict__ xi_f32,
                                 bf16* __restrict__ xi_bf16) {
    int idx = blockIdx.x * blockDim.x + threadIdx.x;   // SEQ * D_INNER
    if (idx >= SEQ * D_INNER) return;
    int t = idx / D_INNER, c = idx % D_INNER;
    float acc = cb[c];
#pragma unroll
    for (int j = 0; j < D_CONV; ++j) {
        int tt = t + j - (D_CONV - 1);
        if (tt >= 0) acc += cw[c * D_CONV + j] * xz[(size_t)tt * D_XZ + c];
    }
    float s = silu_f(acc);
    xi_f32[idx] = s;
    xi_bf16[idx] = (bf16)s;
}

// ---------------------------------------------------------------------------
// Selective scan. One thread per channel (1536 total), 16-state recurrence in
// registers, B/C vectors for timestep t staged in LDS. Fuses +D*xi and
// SiLU(z) gating; emits bf16 y for the out_proj WMMA GEMM.
// ---------------------------------------------------------------------------
__global__ void scan_kernel(const float* __restrict__ dt,
                            const float* __restrict__ xdbc,
                            const float* __restrict__ xi,
                            const float* __restrict__ xz,
                            const float* __restrict__ Alog,
                            const float* __restrict__ Dp,
                            bf16* __restrict__ y_bf16) {
    const int i = blockIdx.x * blockDim.x + threadIdx.x;  // channel
    float Ai[D_STATE];
#pragma unroll
    for (int s = 0; s < D_STATE; ++s)
        Ai[s] = -expf(Alog[(size_t)i * D_STATE + s]);
    const float Di = Dp[i];

    float h[D_STATE];
#pragma unroll
    for (int s = 0; s < D_STATE; ++s) h[s] = 0.0f;

    __shared__ float sBC[2 * D_STATE];   // [0..15]=B(t), [16..31]=C(t)

    for (int t = 0; t < SEQ; ++t) {
        if (threadIdx.x < 2 * D_STATE)
            sBC[threadIdx.x] = xdbc[(size_t)t * XDBC + DT_RANK + threadIdx.x];
        __syncthreads();

        const float dti = dt[(size_t)t * D_INNER + i];
        const float xit = xi[(size_t)t * D_INNER + i];
        const float du  = dti * xit;
        float y = 0.0f;
#pragma unroll
        for (int s = 0; s < D_STATE; ++s) {
            float dA = expf(dti * Ai[s]);
            h[s] = dA * h[s] + du * sBC[s];
            y += sBC[D_STATE + s] * h[s];
        }
        const float zv = xz[(size_t)t * D_XZ + D_INNER + i];
        y = (y + Di * xit) * silu_f(zv);
        y_bf16[(size_t)t * D_INNER + i] = (bf16)y;
        __syncthreads();
    }
}

// ---------------------------------------------------------------------------
// Host launcher
// ---------------------------------------------------------------------------
static inline size_t align256(size_t x) { return (x + 255) & ~(size_t)255; }

extern "C" void kernel_launch(void* const* d_in, const int* in_sizes, int n_in,
                              void* d_out, int out_size, void* d_ws, size_t ws_size,
                              hipStream_t stream) {
    (void)in_sizes; (void)n_in; (void)out_size; (void)ws_size;

    const float* hidden     = (const float*)d_in[0];
    const float* norm_w     = (const float*)d_in[1];
    const float* in_proj_w  = (const float*)d_in[2];
    const float* conv_w     = (const float*)d_in[3];
    const float* conv_b     = (const float*)d_in[4];
    const float* x_proj_w   = (const float*)d_in[5];
    const float* dt_proj_w  = (const float*)d_in[6];
    const float* dt_proj_b  = (const float*)d_in[7];
    const float* A_log      = (const float*)d_in[8];
    const float* D_param    = (const float*)d_in[9];
    const float* out_proj_w = (const float*)d_in[10];
    const float* norm_f_w   = (const float*)d_in[11];

    // Workspace carve-out (≈44 MB)
    char* base = (char*)d_ws;
    size_t off = 0;
    auto carve = [&](size_t bytes) -> void* {
        void* r = base + off;
        off = align256(off + bytes);
        return r;
    };
    bf16*  wInB  = (bf16*) carve((size_t)D_XZ * D_MODEL * 2);       // 3072x768
    bf16*  wXB   = (bf16*) carve((size_t)XDBC * D_INNER * 2);       // 80x1536
    bf16*  wDtB  = (bf16*) carve((size_t)D_INNER * DT_PAD * 2);     // 1536x64
    bf16*  wOutB = (bf16*) carve((size_t)D_MODEL * D_INNER * 2);    // 768x1536
    float* xRes  = (float*)carve((size_t)SEQ * D_MODEL * 4);
    bf16*  hB    = (bf16*) carve((size_t)SEQ * D_MODEL * 2);
    float* xz    = (float*)carve((size_t)SEQ * D_XZ * 4);
    float* xiF   = (float*)carve((size_t)SEQ * D_INNER * 4);
    bf16*  xiB   = (bf16*) carve((size_t)SEQ * D_INNER * 2);
    float* xdbc  = (float*)carve((size_t)SEQ * XDBC * 4);
    bf16*  dtpad = (bf16*) carve((size_t)SEQ * DT_PAD * 2);
    float* dtF   = (float*)carve((size_t)SEQ * D_INNER * 4);
    bf16*  yB    = (bf16*) carve((size_t)SEQ * D_INNER * 2);

    // residual accumulator starts as the input
    hipMemcpyAsync(xRes, hidden, (size_t)SEQ * D_MODEL * sizeof(float),
                   hipMemcpyDeviceToDevice, stream);

    const dim3 blk(256);

    for (int l = 0; l < N_LAYER; ++l) {
        // --- per-layer weight conversion to bf16 ---
        {
            int n1 = D_XZ * D_MODEL;        // 2359296
            f32_to_bf16_kernel<<<(n1 + 255) / 256, blk, 0, stream>>>(
                in_proj_w + (size_t)l * n1, wInB, n1);
            int n2 = XDBC * D_INNER;        // 122880
            f32_to_bf16_kernel<<<(n2 + 255) / 256, blk, 0, stream>>>(
                x_proj_w + (size_t)l * n2, wXB, n2);
            int n3 = D_INNER * DT_PAD;      // 98304 (padded)
            dtw_pad_kernel<<<(n3 + 255) / 256, blk, 0, stream>>>(
                dt_proj_w + (size_t)l * D_INNER * DT_RANK, wDtB, n3);
            int n4 = D_MODEL * D_INNER;     // 1179648
            f32_to_bf16_kernel<<<(n4 + 255) / 256, blk, 0, stream>>>(
                out_proj_w + (size_t)l * n4, wOutB, n4);
        }

        // --- h = RMSNorm(x) (bf16 for WMMA A operand) ---
        rmsnorm_kernel<bf16><<<SEQ, blk, 0, stream>>>(
            xRes, norm_w + (size_t)l * D_MODEL, hB, D_MODEL);

        // --- xz = h @ in_proj_w^T : [1024x3072] = [1024x768]x[3072x768]^T ---
        gemm_wmma_kernel<4, 0><<<dim3(D_XZ / 64, SEQ / 128), blk, 0, stream>>>(
            hB, wInB, xz, nullptr, SEQ, D_XZ, D_MODEL, D_MODEL, D_MODEL, D_XZ);

        // --- depthwise causal conv + SiLU on xi half ---
        conv_silu_kernel<<<(SEQ * D_INNER) / 256, blk, 0, stream>>>(
            xz, conv_w + (size_t)l * D_INNER * D_CONV,
            conv_b + (size_t)l * D_INNER, xiF, xiB);

        // --- xdbc = xi @ x_proj_w^T : [1024x80] = [1024x1536]x[80x1536]^T ---
        gemm_wmma_kernel<5, 0><<<dim3(1, SEQ / 128), blk, 0, stream>>>(
            xiB, wXB, xdbc, nullptr, SEQ, XDBC, D_INNER, D_INNER, D_INNER, XDBC);

        // --- pad dt rank 48 -> 64 as bf16 ---
        dt_pad_act_kernel<<<(SEQ * DT_PAD) / 256, blk, 0, stream>>>(xdbc, dtpad);

        // --- dt = softplus(dt_lowrank @ dt_proj_w^T + b) : [1024x1536] ---
        gemm_wmma_kernel<4, 2><<<dim3(D_INNER / 64, SEQ / 128), blk, 0, stream>>>(
            dtpad, wDtB, dtF, dt_proj_b + (size_t)l * D_INNER,
            SEQ, D_INNER, DT_PAD, DT_PAD, DT_PAD, D_INNER);

        // --- selective scan (fuses +D*xi and SiLU(z) gate) -> y bf16 ---
        scan_kernel<<<D_INNER / 256, blk, 0, stream>>>(
            dtF, xdbc, xiF, xz,
            A_log + (size_t)l * D_INNER * D_STATE,
            D_param + (size_t)l * D_INNER, yB);

        // --- x += y @ out_proj_w^T : [1024x768], in-place residual epilogue ---
        gemm_wmma_kernel<4, 1><<<dim3(D_MODEL / 64, SEQ / 128), blk, 0, stream>>>(
            yB, wOutB, xRes, xRes, SEQ, D_MODEL, D_INNER, D_INNER, D_INNER, D_MODEL);
    }

    // --- final RMSNorm -> d_out (f32) ---
    rmsnorm_kernel<float><<<SEQ, blk, 0, stream>>>(
        xRes, norm_f_w, (float*)d_out, D_MODEL);
}
